// SSM_69630009803018
// MI455X (gfx1250) — matrix operand order repaired and checked by reading, not compiled, not run on hardware
//
#include <hip/hip_runtime.h>
#include <math.h>

// ---------- problem constants (baked in for strength-reduced addressing) ----------
#define CC   1024            // feature dim
#define TT   1024            // sequence length
#define BB   8               // batch
#define MMR  (BB * TT)       // 8192 rows
#define LDA2 (2 * CC)        // concatenated-K leading dim (2048)

// ---------- types ----------
typedef __attribute__((ext_vector_type(16))) _Float16 v16h;
typedef __attribute__((ext_vector_type(8)))  _Float16 v8h;
typedef __attribute__((ext_vector_type(8)))  float    v8f;
typedef __attribute__((ext_vector_type(4)))  unsigned u32x4;
typedef __attribute__((ext_vector_type(4)))  int      i32x4;
typedef __attribute__((ext_vector_type(8)))  int      i32x8;

// ---------- WMMA helpers (CDNA5 gfx1250, wave32) ----------
__device__ inline v8f wmma_f16(v16h a, v16h b, v8f c) {
  return __builtin_amdgcn_wmma_f32_16x16x32_f16(false, a, false, b, (short)0, c, false, false);
}

// A fragment from a per-lane base pointer (row-major, contiguous K):
// elems 0..15 = K +0..15 (+16 for hi half-wave)
__device__ inline v16h ld_a(const _Float16* p) {
  return *(const v16h*)p;
}

// B fragment: contiguous 16 halves per lane
__device__ inline v16h ld_b(const _Float16* p) { return *(const v16h*)p; }

// ---------- Tensor Data Mover: stage a 2D f16 tile into LDS ----------
// D# per CDNA5 ISA 8.3/8.4: group0 = {flags, lds_addr, global_addr, type=2},
// group1 = {data_size=2B, tensor_dim0/1, tile_dim0/1, dim0_stride}; groups 2/3 zero (2D).
__device__ inline void tdm_load_2d_to_lds(unsigned lds_addr, const void* gaddr,
                                          unsigned dim0, unsigned dim1,
                                          unsigned tile0, unsigned tile1,
                                          unsigned stride0) {
  unsigned long long ga = (unsigned long long)(size_t)gaddr;
  u32x4 g0;
  g0[0] = 1u;                                            // count=1, user descriptor
  g0[1] = lds_addr;                                      // lds_addr @ bits[63:32]
  g0[2] = (unsigned)(ga & 0xFFFFFFFFu);                  // global_addr[31:0]
  g0[3] = (unsigned)((ga >> 32) & 0x01FFFFFFu) | (2u << 30);  // addr[56:32] | type=2
  i32x8 g1;
  g1[0] = (int)(1u << 16);                               // data_size=1 (2 bytes)
  g1[1] = (int)((dim0 & 0xFFFFu) << 16);                 // tensor_dim0[15:0] @ [63:48]
  g1[2] = (int)(((dim0 >> 16) & 0xFFFFu) | ((dim1 & 0xFFFFu) << 16));
  g1[3] = (int)(((dim1 >> 16) & 0xFFFFu) | ((tile0 & 0xFFFFu) << 16));
  g1[4] = (int)(tile1 & 0xFFFFu);                        // tile_dim1 | tile_dim2=0
  g1[5] = (int)stride0;                                  // tensor_dim0_stride[31:0]
  g1[6] = 0;
  g1[7] = 0;
  i32x4 z4 = {0, 0, 0, 0};
#if defined(__clang_major__) && (__clang_major__ >= 23)
  i32x8 z8 = {0, 0, 0, 0, 0, 0, 0, 0};
  __builtin_amdgcn_tensor_load_to_lds(g0, g1, z4, z4, z8, 0);
#else
  __builtin_amdgcn_tensor_load_to_lds(g0, g1, z4, z4, 0);
#endif
}

// ---------- ordered-int float min/max keys for global atomics ----------
__device__ inline unsigned fkey(float f) {
  unsigned u = __float_as_uint(f);
  return u ^ (unsigned)(((int)u >> 31) | 0x80000000);
}
__device__ inline float fkey_inv(unsigned k) {
  unsigned u = (k & 0x80000000u) ? (k ^ 0x80000000u) : ~k;
  return __uint_as_float(u);
}

// ---------- elementwise / prep kernels ----------
__global__ void f32_to_f16_kernel(const float* __restrict__ in, _Float16* __restrict__ out, int n) {
  int i = blockIdx.x * blockDim.x + threadIdx.x;
  if (i < n) out[i] = (_Float16)in[i];
}

// pack one polar result into half of the concatenated B operand [CC][2CC]
__global__ void pack_ow_kernel(const float* __restrict__ U, _Float16* __restrict__ owcat, int koff) {
  int idx = blockIdx.x * blockDim.x + threadIdx.x;
  int n = idx >> 10, k = idx & (CC - 1);
  owcat[(long)n * LDA2 + koff + k] = (_Float16)U[idx];
}

__device__ inline unsigned hash_u32(unsigned x) {
  x ^= x >> 17; x *= 0xed5ad4bbu; x ^= x >> 11; x *= 0xac4c1b51u;
  x ^= x >> 15; x *= 0x31848babu; x ^= x >> 14; return x;
}

__global__ void gen_probe_kernel(float* __restrict__ p1, float* __restrict__ p2) {
  int i = blockIdx.x * blockDim.x + threadIdx.x;
  if (i >= CC) return;
  unsigned h1 = hash_u32(0x65000000u + i), h2 = hash_u32(0x66000000u + i);
  float u1 = ((float)h1 + 1.0f) * 2.3283064e-10f;
  float u2 = (float)h2 * 2.3283064e-10f;
  p1[i] = sqrtf(-2.f * logf(u1)) * __cosf(6.2831853f * u2);
  unsigned h3 = hash_u32(0x67000000u + i), h4 = hash_u32(0x68000000u + i);
  float u3 = ((float)h3 + 1.0f) * 2.3283064e-10f;
  float u4 = (float)h4 * 2.3283064e-10f;
  p2[i] = sqrtf(-2.f * logf(u3)) * __cosf(6.2831853f * u4);
}

// per-row dot products U@v : one wave per row (deterministic, no atomics)
__global__ void matvec_kernel(const float* __restrict__ U, const float* __restrict__ v,
                              float* __restrict__ dots) {
  int row  = blockIdx.x * (blockDim.x >> 5) + (threadIdx.x >> 5);
  int lane = threadIdx.x & 31;
  float s = 0.f;
  for (int k = lane; k < CC; k += 32) s += U[(long)row * CC + k] * v[k];
#pragma unroll
  for (int off = 16; off; off >>= 1) s += __shfl_xor(s, off, 32);
  if (lane == 0) dots[row] = s;
}

// fixed-order reduction: s = sqrt(sum dots^2) + 1e-8
__global__ void reduce_sqnorm_kernel(const float* __restrict__ dots, float* __restrict__ s) {
  __shared__ float sm[256];
  float t = 0.f;
  for (int i = threadIdx.x; i < CC; i += 256) { float d = dots[i]; t += d * d; }
  sm[threadIdx.x] = t;
  __syncthreads();
  for (int off = 128; off; off >>= 1) {
    if ((int)threadIdx.x < off) sm[threadIdx.x] += sm[threadIdx.x + off];
    __syncthreads();
  }
  if (threadIdx.x == 0) *s = sqrtf(sm[0]) + 1e-8f;
}

// U' = 0.5*U + I/s - U^T/(2 s^2)
__global__ void polar_update_kernel(const float* __restrict__ U, const float* __restrict__ sp,
                                    float* __restrict__ Un) {
  int idx = blockIdx.x * blockDim.x + threadIdx.x;
  int i = idx >> 10, j = idx & (CC - 1);
  float s = *sp;
  float inv = 1.f / s;
  Un[idx] = 0.5f * U[idx] + ((i == j) ? inv : 0.f) - U[(long)j * CC + i] * (0.5f * inv * inv);
}

// init Acat ([16][2CC]: rows 0..7 = identity | q staged later; rows 8..15 zero) and minmax
__global__ void scan_init_kernel(_Float16* __restrict__ Acat, const float* __restrict__ identity,
                                 unsigned* __restrict__ mm) {
  int i = blockIdx.x * blockDim.x + threadIdx.x;
  if (i < 16 * LDA2) {
    int b = i / LDA2, c = i % LDA2;
    _Float16 val = (_Float16)0.f;
    if (b < 8 && c < CC) val = (_Float16)identity[c];
    Acat[i] = val;
  }
  if (i < 16 * TT) {
    mm[i] = ((i & 15) < 8) ? 0xFFFFFFFFu : 0u;
  }
}

// copy q_0 into the second K-half of Acat
__global__ void stage_q0_kernel(const _Float16* __restrict__ q, _Float16* __restrict__ Acat) {
  int idx = blockIdx.x * blockDim.x + threadIdx.x;
  if (idx >= 8 * CC) return;
  int b = idx >> 10, c = idx & (CC - 1);
  Acat[(long)b * LDA2 + CC + c] = q[(long)b * TT * CC + c];
}

// ---------- main WMMA GEMM: D = A(8192xK f16) * B^T (B stored [N][K] f16), f32 acc ----------
// per-wave tile 32(M)x64(N); 8 independent accumulators; unconditional prefetch
template <int OUT_F16>
__global__ __launch_bounds__(256) void wmma_gemm_nt(const _Float16* __restrict__ A,
                                                    const _Float16* __restrict__ B,
                                                    void* __restrict__ Cout) {
  constexpr int N = CC, K = CC;
  int wid = threadIdx.x >> 5, lane = threadIdx.x & 31;
  int gw = blockIdx.x * 8 + wid;
  int tm = gw >> 4;                  // 16 = N/64 tiles across
  int tn = gw & 15;
  int m0 = tm << 5, n0 = tn << 6;
  int mrow = lane & 15, half = lane >> 4;

  const _Float16* pa0 = A + (long)(m0 + mrow) * K + (half << 4);
  const _Float16* pa1 = pa0 + 16 * K;
  const _Float16* pb0 = B + (long)(n0 + mrow) * K + (half << 4);
  const _Float16* pb1 = pb0 + 16 * K;
  const _Float16* pb2 = pb0 + 32 * K;
  const _Float16* pb3 = pb0 + 48 * K;

  v8f acc[2][4];
#pragma unroll
  for (int i = 0; i < 2; i++)
#pragma unroll
    for (int j = 0; j < 4; j++)
#pragma unroll
      for (int r = 0; r < 8; r++) acc[i][j][r] = 0.f;

#pragma unroll 4
  for (int k0 = 0; k0 < K; k0 += 32) {
    __builtin_prefetch((const void*)(pa0 + k0 + 256), 0, 1);
    __builtin_prefetch((const void*)(pa1 + k0 + 256), 0, 1);
    __builtin_prefetch((const void*)(pb0 + k0 + 256), 0, 1);
    __builtin_prefetch((const void*)(pb1 + k0 + 256), 0, 1);
    __builtin_prefetch((const void*)(pb2 + k0 + 256), 0, 1);
    __builtin_prefetch((const void*)(pb3 + k0 + 256), 0, 1);
    v16h a0 = ld_a(pa0 + k0);
    v16h a1 = ld_a(pa1 + k0);
    v16h b0 = ld_b(pb0 + k0);
    v16h b1 = ld_b(pb1 + k0);
    v16h b2 = ld_b(pb2 + k0);
    v16h b3 = ld_b(pb3 + k0);
    acc[0][0] = wmma_f16(a0, b0, acc[0][0]);
    acc[1][0] = wmma_f16(a1, b0, acc[1][0]);
    acc[0][1] = wmma_f16(a0, b1, acc[0][1]);
    acc[1][1] = wmma_f16(a1, b1, acc[1][1]);
    acc[0][2] = wmma_f16(a0, b2, acc[0][2]);
    acc[1][2] = wmma_f16(a1, b2, acc[1][2]);
    acc[0][3] = wmma_f16(a0, b3, acc[0][3]);
    acc[1][3] = wmma_f16(a1, b3, acc[1][3]);
  }

#pragma unroll
  for (int i = 0; i < 2; i++)
#pragma unroll
    for (int j = 0; j < 4; j++)
#pragma unroll
      for (int r = 0; r < 8; r++) {
        long m = m0 + (i << 4) + r + (half << 3);
        long n = n0 + (j << 4) + mrow;
        if (OUT_F16) ((_Float16*)Cout)[m * N + n] = (_Float16)acc[i][j][r];
        else         ((float*)Cout)[m * N + n] = acc[i][j][r];
      }
}

// ---------- scan step: z = [state|q_t] @ [ow1|ow2]^T over K=2CC ----------
// A staged to LDS by the Tensor Data Mover. Lanes 8..15 and 24..31 load duplicates of rows
// 0..7 (no zero-select): WMMA output rows depend only on matching A rows, and
// rows 8..15 of D are never stored nor reduced, so duplicates are harmless.
__global__ __launch_bounds__(256) void combine_step_kernel(
    const _Float16* __restrict__ Acat,    // [16][2CC] f16 (rows 0..7 staged to LDS)
    const _Float16* __restrict__ owcat,   // [CC][2CC] f16
    float* __restrict__ zbuf,             // [8][CC]
    unsigned* __restrict__ mm) {          // this step's [min x8, max x8] keys
  __shared__ _Float16 sh_a[8 * LDA2];     // 32 KB
  __shared__ float red[4][2][256];        // 8 KB
  int wid  = threadIdx.x >> 5;
  int lane = threadIdx.x & 31;

  if (wid == 0) {  // one TDM op per workgroup; TENSORcnt is per-wave
    tdm_load_2d_to_lds((unsigned)(size_t)(void*)sh_a, (const void*)Acat,
                       LDA2, 8, LDA2, 8, LDA2);
    __builtin_amdgcn_s_wait_tensorcnt(0);
  }
  __syncthreads();

  int pair = wid >> 1;                    // 0..3, two 16-col tiles per pair
  int kh   = wid & 1;                     // split-K half
  int tbase = blockIdx.x * 8 + pair * 2;
  int kbeg = kh * CC;

  const _Float16* sa  = sh_a + (lane & 7) * LDA2 + ((lane >> 4) << 4) + kbeg;
  const _Float16* pb0 = owcat + (long)((tbase << 4) + (lane & 15)) * LDA2 + ((lane >> 4) << 4) + kbeg;
  const _Float16* pb1 = pb0 + 16 * LDA2;

  v8f acc[2];
#pragma unroll
  for (int j = 0; j < 2; j++)
#pragma unroll
    for (int r = 0; r < 8; r++) acc[j][r] = 0.f;

#pragma unroll 4
  for (int kk = 0; kk < CC; kk += 32) {
    __builtin_prefetch((const void*)(pb0 + kk + 256), 0, 1);
    __builtin_prefetch((const void*)(pb1 + kk + 256), 0, 1);
    v16h a  = ld_a(sa + kk);
    v16h b0 = ld_b(pb0 + kk);
    v16h b1 = ld_b(pb1 + kk);
    acc[0] = wmma_f16(a, b0, acc[0]);
    acc[1] = wmma_f16(a, b1, acc[1]);
  }

  if (kh == 1) {
#pragma unroll
    for (int j = 0; j < 2; j++)
#pragma unroll
      for (int r = 0; r < 8; r++) red[pair][j][lane * 8 + r] = acc[j][r];
  }
  __syncthreads();
  if (kh == 0) {
#pragma unroll
    for (int j = 0; j < 2; j++)
#pragma unroll
      for (int r = 0; r < 8; r++) acc[j][r] += red[pair][j][lane * 8 + r];

    int nn = lane & 15, half = lane >> 4;
#pragma unroll
    for (int j = 0; j < 2; j++) {
      int n0 = (tbase + j) << 4;
#pragma unroll
      for (int r = 0; r < 8; r++) {
        if (half == 0) zbuf[(long)r * CC + n0 + nn] = acc[j][r];  // rows 0..7 only
        float mn = acc[j][r], mx = acc[j][r];
#pragma unroll
        for (int off = 8; off; off >>= 1) {  // stays within lanes 0..15 (valid rows)
          mn = fminf(mn, __shfl_xor(mn, off, 32));
          mx = fmaxf(mx, __shfl_xor(mx, off, 32));
        }
        if (lane == 0) {
          atomicMin(&mm[r], fkey(mn));
          atomicMax(&mm[8 + r], fkey(mx));
        }
      }
    }
  }
}

// range-normalize z, update state half of Acat, stage q_{t+1}, emit prod = norm(z)*v
__global__ void normalize_step_kernel(const float* __restrict__ zbuf,
                                      const unsigned* __restrict__ mm,
                                      const float* __restrict__ vbase,
                                      const _Float16* __restrict__ qnext,
                                      _Float16* __restrict__ Acat,
                                      _Float16* __restrict__ prodbase) {
  int idx = blockIdx.x * blockDim.x + threadIdx.x;  // [0, 8*CC)
  if (idx >= 8 * CC) return;
  int b = idx >> 10, c = idx & (CC - 1);
  constexpr long ldr = (long)TT * CC;
  float zmin = fkey_inv(mm[b]);
  float zmax = fkey_inv(mm[8 + b]);
  float z = zbuf[idx];
  float nrm = (z - zmin) / (zmax - zmin + 1e-8f) * 2.f - 1.f;
  Acat[(long)b * LDA2 + c] = (_Float16)nrm;
  if (qnext) Acat[(long)b * LDA2 + CC + c] = qnext[(long)b * ldr + c];
  prodbase[(long)b * ldr + c] = (_Float16)(nrm * vbase[(long)b * ldr + c]);
}

// ---------- host orchestration ----------
extern "C" void kernel_launch(void* const* d_in, const int* in_sizes, int n_in,
                              void* d_out, int out_size, void* d_ws, size_t ws_size,
                              hipStream_t stream) {
  (void)in_sizes; (void)n_in; (void)out_size; (void)ws_size;
  const int C = CC, T = TT, M = MMR;

  const float* x        = (const float*)d_in[0];
  const float* Wq       = (const float*)d_in[1];
  const float* Wv       = (const float*)d_in[2];
  const float* Wc       = (const float*)d_in[3];
  const float* identity = (const float*)d_in[4];
  const float* P1       = (const float*)d_in[5];
  const float* P2       = (const float*)d_in[6];
  float* out = (float*)d_out;

  char* ws = (char*)d_ws;
  size_t off = 0;
  auto take = [&](size_t bytes) -> char* {   // +4KB pad so unconditional prefetch stays in-bounds
    char* p = ws + off;
    off = (off + bytes + 4096 + 255) & ~(size_t)255;
    return p;
  };

  _Float16* x_h    = (_Float16*)take((size_t)M * C * 2);
  _Float16* q_h    = (_Float16*)take((size_t)M * C * 2);
  float*    v_f    = (float*)   take((size_t)M * C * 4);
  _Float16* p_h    = (_Float16*)take((size_t)M * C * 2);
  _Float16* wq_h   = (_Float16*)take((size_t)C * C * 2);
  _Float16* wv_h   = (_Float16*)take((size_t)C * C * 2);
  _Float16* wc_h   = (_Float16*)take((size_t)C * C * 2);
  _Float16* owcat  = (_Float16*)take((size_t)C * LDA2 * 2);
  float*    Ua     = (float*)   take((size_t)C * C * 4);
  float*    Ub     = (float*)   take((size_t)C * C * 4);
  float*    probe1 = (float*)   take((size_t)C * 4);
  float*    probe2 = (float*)   take((size_t)C * 4);
  float*    dots   = (float*)   take((size_t)C * 4);
  float*    svals  = (float*)   take(4 * 4);
  _Float16* Acat   = (_Float16*)take((size_t)16 * LDA2 * 2);
  float*    zbuf   = (float*)   take((size_t)8 * C * 4);
  unsigned* mm     = (unsigned*)take((size_t)T * 16 * 4);

  // ---- init ----
  scan_init_kernel<<<128, 256, 0, stream>>>(Acat, identity, mm);
  gen_probe_kernel<<<(C + 255) / 256, 256, 0, stream>>>(probe1, probe2);

  // ---- pack f32 operands to f16 for the WMMA path ----
  f32_to_f16_kernel<<<(M * C) / 256, 256, 0, stream>>>(x, x_h, M * C);
  f32_to_f16_kernel<<<(C * C) / 256, 256, 0, stream>>>(Wq, wq_h, C * C);
  f32_to_f16_kernel<<<(C * C) / 256, 256, 0, stream>>>(Wv, wv_h, C * C);
  f32_to_f16_kernel<<<(C * C) / 256, 256, 0, stream>>>(Wc, wc_h, C * C);

  // ---- fast_polar(P1, v1) / fast_polar(P2, v2), 2 iterations, deterministic norm ----
  const float* Ps[2]     = {P1, P2};
  const float* probes[2] = {probe1, probe2};
  for (int p = 0; p < 2; p++) {
    float* s0 = svals + 2 * p;
    float* s1 = svals + 2 * p + 1;
    matvec_kernel<<<C / 8, 256, 0, stream>>>(Ps[p], probes[p], dots);
    reduce_sqnorm_kernel<<<1, 256, 0, stream>>>(dots, s0);
    polar_update_kernel<<<(C * C) / 256, 256, 0, stream>>>(Ps[p], s0, Ua);
    matvec_kernel<<<C / 8, 256, 0, stream>>>(Ua, probes[p], dots);
    reduce_sqnorm_kernel<<<1, 256, 0, stream>>>(dots, s1);
    polar_update_kernel<<<(C * C) / 256, 256, 0, stream>>>(Ua, s1, Ub);
    pack_ow_kernel<<<(C * C) / 256, 256, 0, stream>>>(Ub, owcat, p * C);
  }

  // ---- q = x @ Wq^T (f16 out), v = x @ Wv^T (f32 out) ----
  int gemm_blocks = (M / 32) * (C / 64) / 8;   // 512
  wmma_gemm_nt<1><<<gemm_blocks, 256, 0, stream>>>(x_h, wq_h, (void*)q_h);
  wmma_gemm_nt<0><<<gemm_blocks, 256, 0, stream>>>(x_h, wv_h, (void*)v_f);
  stage_q0_kernel<<<(8 * C) / 256, 256, 0, stream>>>(q_h, Acat);

  // ---- serial scan: combine (TDM-staged A, WMMA, split-K, LDS reduce) + range-norm ----
  for (int t = 0; t < T; t++) {
    combine_step_kernel<<<C / 16 / 8, 256, 0, stream>>>(
        Acat, owcat, zbuf, mm + (size_t)t * 16);
    const _Float16* qnext = (t + 1 < T) ? (q_h + (size_t)(t + 1) * C) : (const _Float16*)nullptr;
    normalize_step_kernel<<<(8 * C) / 256, 256, 0, stream>>>(
        zbuf, mm + (size_t)t * 16, v_f + (size_t)t * C,
        qnext, Acat, p_h + (size_t)t * C);
  }

  // ---- Y = prod @ Wc^T (f32 out to d_out) ----
  wmma_gemm_nt<0><<<gemm_blocks, 256, 0, stream>>>(p_h, wc_h, (void*)out);
}
